// ConvolutionUpdateFeature_64776696758988
// MI455X (gfx1250) — matrix operand order repaired and checked by reading, not compiled, not run on hardware
//
#include <hip/hip_runtime.h>

// ---------------------------------------------------------------------------
// MI455X (gfx1250, wave32) implementation of the FermiNet-style
// ConvolutionUpdateFeature. bf16 WMMA (v_wmma_f32_16x16x32_bf16) with fp32
// accumulation; spin-block masking is exploited to skip half of the
// edges_same / edges_anti traffic (memory floor ~9us @ 23.3 TB/s).
// ---------------------------------------------------------------------------

typedef __attribute__((ext_vector_type(16))) __bf16 v16bf;
typedef __attribute__((ext_vector_type(8)))  float  v8f;

#define NBATCH 1024
#define SA 272   // LDS K-stride (bf16 elems) for 256-K node weights (32B aligned rows)
#define SE 48    // LDS K-stride (bf16 elems) for 32-K edge weights  (32B aligned rows)

__device__ __forceinline__ float fast_tanh(float x) {
#if __has_builtin(__builtin_amdgcn_tanhf)
    return __builtin_amdgcn_tanhf(x);                 // v_tanh_f32
#elif __has_builtin(__builtin_amdgcn_tanh_f32)
    return __builtin_amdgcn_tanh_f32(x);
#else
    float e = __builtin_amdgcn_exp2f(2.885390081777927f * x);  // exp(2x)
    return 1.0f - 2.0f / (e + 1.0f);
#endif
}

// Load 16 K-values of the A matrix for this lane (fp32 -> bf16).
// Per ISA: lanes 0-15 hold K[0..7],K[16..23]; lanes 16-31 hold K[8..15],K[24..31]
// within a 32-wide K window -> caller passes p = row + 32*s + hi*8.
__device__ __forceinline__ v16bf cvt_a(const float* __restrict__ p) {
    v8f lo = *(const v8f*)(p);
    v8f hv = *(const v8f*)(p + 16);
    v16bf a;
#pragma unroll
    for (int q = 0; q < 8; ++q) { a[q] = (__bf16)lo[q]; a[q + 8] = (__bf16)hv[q]; }
    return a;
}

// B operand: lane holds column n, K[hi*8 .. hi*8+7], K[16+hi*8 .. 16+hi*8+7] (two 16B LDS loads).
__device__ __forceinline__ v16bf load_b(const __bf16* __restrict__ Wt, int stride,
                                        int n, int kbase, int hi) {
    typedef __attribute__((ext_vector_type(8))) __bf16 v8bf;
    const __bf16* q = Wt + n * stride + kbase + hi * 8;
    v8bf lo = *(const v8bf*)(q);
    v8bf hv = *(const v8bf*)(q + 16);
    return __builtin_shufflevector(lo, hv, 0, 1, 2, 3, 4, 5, 6, 7, 8, 9, 10, 11, 12, 13, 14, 15);
}

// ---------------------------------------------------------------------------
// Kernel A: out[r, 0..63] = tanh(X[r, 0..255] @ W[256,64] + b)    (row tiles of 16)
// ---------------------------------------------------------------------------
__global__ __launch_bounds__(256) void hx_mlp_kernel(
    const float* __restrict__ X, const float* __restrict__ W,
    const float* __restrict__ bias, float* __restrict__ out, int ntiles)
{
    __shared__ __bf16 Wt[64 * SA];
    __shared__ float  bsm[64];
    for (int idx = threadIdx.x; idx < 256 * 64; idx += 256) {
        int k = idx >> 6, n = idx & 63;           // W is [k][n] row-major
        Wt[n * SA + k] = (__bf16)W[idx];
    }
    if (threadIdx.x < 64) bsm[threadIdx.x] = bias[threadIdx.x];
    __syncthreads();

    const int lane = threadIdx.x & 31;
    const int l16  = lane & 15;
    const int hi   = lane >> 4;
    const int koff = hi * 8;
    const int gw   = blockIdx.x * (blockDim.x >> 5) + (threadIdx.x >> 5);
    const int nw   = gridDim.x * (blockDim.x >> 5);

    for (int t = gw; t < ntiles; t += nw) {
        const float* xr = X + (size_t)(t * 16 + l16) * 256;
        v8f acc[4] = {};
#pragma unroll
        for (int s = 0; s < 8; ++s) {             // K = 256 in 8 steps of 32
            v16bf a = cvt_a(xr + 32 * s + koff);
#pragma unroll
            for (int nt = 0; nt < 4; ++nt) {
                v16bf bm = load_b(Wt, SA, nt * 16 + l16, 32 * s, hi);
                acc[nt] = __builtin_amdgcn_wmma_f32_16x16x32_bf16(
                    false, a, false, bm, (short)0, acc[nt], false, false);
            }
        }
        const int rb = hi * 8;                    // D rows: lanes>=16 own M=8..15
#pragma unroll
        for (int nt = 0; nt < 4; ++nt) {
            int d = nt * 16 + l16;
            float bb = bsm[d];
#pragma unroll
            for (int r = 0; r < 8; ++r)
                out[(size_t)(t * 16 + rb + r) * 64 + d] = fast_tanh(acc[nt][r] + bb);
        }
    }
}

// ---------------------------------------------------------------------------
// Kernel B helpers
// ---------------------------------------------------------------------------
// One masked ee contribution for receiver i over its 16-wide j block.
__device__ __forceinline__ void ee_part(
    const float* __restrict__ ebase,              // edges + ((b*32+i)*32 + jbase)*32
    const __bf16* __restrict__ Wt, const float* __restrict__ bias,
    const float* __restrict__ hx, int jbase, int idiag,
    int l16, int hi, float res[4])
{
    const int koff = hi * 8;
    v16bf a = cvt_a(ebase + l16 * 32 + koff);     // rows = 16 senders j
    const int rb = hi * 8;
#pragma unroll
    for (int nt = 0; nt < 4; ++nt) {
        v16bf bm = load_b(Wt, SE, nt * 16 + l16, 0, hi);
        v8f acc = {};
        acc = __builtin_amdgcn_wmma_f32_16x16x32_bf16(
            false, a, false, bm, (short)0, acc, false, false);
        int d = nt * 16 + l16;
        float bb = bias[d];
        float sum = 0.f;
#pragma unroll
        for (int r = 0; r < 8; ++r) {
            int j = jbase + rb + r;
            float v = fast_tanh(acc[r] + bb);
            float h = (j == idiag) ? 0.f : hx[j * 64 + d];
            sum += v * h;
        }
        sum += __shfl_xor(sum, 16, 32);           // merge the two D row-halves
        res[nt] = sum;
    }
}

// ne contribution for the receiver pair (i0, i0+1): rows 0-7 = i0's nuclei,
// rows 8-15 = i1's nuclei -> lanes 0-15 own i0, lanes 16-31 own i1.
__device__ __forceinline__ void ne_part(
    const float* __restrict__ enb,                // edges_ne + b*32*8*32
    int i0, const __bf16* __restrict__ Wt, const float* __restrict__ bias,
    const float* __restrict__ hx, int l16, int hi, float res[4])
{
    const int koff  = hi * 8;
    const int i_row = i0 + (l16 >> 3);
    const int n_row = l16 & 7;
    v16bf a = cvt_a(enb + ((size_t)i_row * 8 + n_row) * 32 + koff);
#pragma unroll
    for (int nt = 0; nt < 4; ++nt) {
        v16bf bm = load_b(Wt, SE, nt * 16 + l16, 0, hi);
        v8f acc = {};
        acc = __builtin_amdgcn_wmma_f32_16x16x32_bf16(
            false, a, false, bm, (short)0, acc, false, false);
        int d = nt * 16 + l16;
        float bb = bias[d];
        float sum = 0.f;
#pragma unroll
        for (int r = 0; r < 8; ++r)
            sum += fast_tanh(acc[r] + bb) * hx[r * 64 + d];
        res[nt] = sum * 0.125f;                   // / n_nuc
    }
}

// ---------------------------------------------------------------------------
// Kernel B: per-batch masked edge convolutions.  1 block = 1 batch, 4 waves,
// wave w handles receivers i = 8w .. 8w+7 (uniform spin half per wave).
// ---------------------------------------------------------------------------
__global__ __launch_bounds__(128) void conv_kernel(
    const float* __restrict__ edges_same, const float* __restrict__ edges_anti,
    const float* __restrict__ edges_ne,
    const float* __restrict__ wS, const float* __restrict__ bS,
    const float* __restrict__ wA, const float* __restrict__ bA,
    const float* __restrict__ wN, const float* __restrict__ bN,
    const float* __restrict__ hxS, const float* __restrict__ hxA,
    const float* __restrict__ hxN,
    float* __restrict__ out_ee, float* __restrict__ out_ne)
{
    __shared__ __bf16 WS[64 * SE], WA[64 * SE], WN[64 * SE];
    __shared__ float  bSs[64], bAs[64], bNs[64];
    __shared__ float  hS[32 * 64], hA[32 * 64], hN[8 * 64];

    const int b = blockIdx.x;
    for (int idx = threadIdx.x; idx < 2048; idx += 128) {
        int k = idx >> 6, n = idx & 63;           // w*_W is [k=32][n=64]
        WS[n * SE + k] = (__bf16)wS[idx];
        WA[n * SE + k] = (__bf16)wA[idx];
        WN[n * SE + k] = (__bf16)wN[idx];
        hS[idx] = hxS[(size_t)b * 2048 + idx];
        hA[idx] = hxA[(size_t)b * 2048 + idx];
    }
    for (int idx = threadIdx.x; idx < 512; idx += 128)
        hN[idx] = hxN[(size_t)b * 512 + idx];
    if (threadIdx.x < 64) {
        bSs[threadIdx.x] = bS[threadIdx.x];
        bAs[threadIdx.x] = bA[threadIdx.x];
        bNs[threadIdx.x] = bN[threadIdx.x];
    }
    __syncthreads();

    const int lane = threadIdx.x & 31;
    const int wid  = threadIdx.x >> 5;
    const int l16  = lane & 15;
    const int hi   = lane >> 4;
    const int ibase = wid * 8;

    const float* esb = edges_same + (size_t)b * 32 * 32 * 32;
    const float* eab = edges_anti + (size_t)b * 32 * 32 * 32;
    const float* enb = edges_ne   + (size_t)b * 32 * 8 * 32;
    float* oe = out_ee + (size_t)b * 32 * 64;
    float* on = out_ne + (size_t)b * 32 * 64;

    for (int ii = 0; ii < 8; ++ii) {
        int i = ibase + ii;
        int jbS = (i < 16) ? 0 : 16;              // same-spin j block
        int jbA = 16 - jbS;                       // anti-spin j block
        float rs[4], ra[4];
        ee_part(esb + ((size_t)i * 32 + jbS) * 32, WS, bSs, hS, jbS, i,  l16, hi, rs);
        ee_part(eab + ((size_t)i * 32 + jbA) * 32, WA, bAs, hA, jbA, -1, l16, hi, ra);
        if (hi == 0) {
#pragma unroll
            for (int nt = 0; nt < 4; ++nt)
                oe[(size_t)i * 64 + nt * 16 + l16] = (rs[nt] + ra[nt]) * (1.0f / 32.0f);
        }
    }

    for (int ip = 0; ip < 4; ++ip) {
        int i0 = ibase + 2 * ip;
        float rn[4];
        ne_part(enb, i0, WN, bNs, hN, l16, hi, rn);
        int iw = i0 + hi;                         // lanes<16 -> i0, lanes>=16 -> i1
#pragma unroll
        for (int nt = 0; nt < 4; ++nt)
            on[(size_t)iw * 64 + nt * 16 + l16] = rn[nt];
    }
}

// ---------------------------------------------------------------------------
extern "C" void kernel_launch(void* const* d_in, const int* in_sizes, int n_in,
                              void* d_out, int out_size, void* d_ws, size_t ws_size,
                              hipStream_t stream) {
    (void)in_sizes; (void)n_in; (void)out_size; (void)ws_size;
    const float* nodes_elec = (const float*)d_in[0];
    const float* nodes_nuc  = (const float*)d_in[1];
    const float* edges_same = (const float*)d_in[2];
    const float* edges_anti = (const float*)d_in[3];
    const float* edges_ne   = (const float*)d_in[4];
    const float* w_same_W = (const float*)d_in[5];
    const float* w_same_b = (const float*)d_in[6];
    const float* w_anti_W = (const float*)d_in[7];
    const float* w_anti_b = (const float*)d_in[8];
    const float* w_ne_W   = (const float*)d_in[9];
    const float* w_ne_b   = (const float*)d_in[10];
    const float* h_same_W = (const float*)d_in[11];
    const float* h_same_b = (const float*)d_in[12];
    const float* h_anti_W = (const float*)d_in[13];
    const float* h_anti_b = (const float*)d_in[14];
    const float* h_ne_W   = (const float*)d_in[15];
    const float* h_ne_b   = (const float*)d_in[16];

    // workspace: hx_same[B*32*64] | hx_anti[B*32*64] | hx_ne[B*8*64]  (~19 MB)
    float* hxS = (float*)d_ws;
    float* hxA = hxS + (size_t)NBATCH * 32 * 64;
    float* hxN = hxA + (size_t)NBATCH * 32 * 64;

    float* out_ee = (float*)d_out;
    float* out_ne = out_ee + (size_t)NBATCH * 32 * 64;

    // hx GEMMs: elec rows = 1024*32 = 32768 (2048 tiles), nuc rows = 8192 (512)
    hx_mlp_kernel<<<128, 256, 0, stream>>>(nodes_elec, h_same_W, h_same_b, hxS, 2048);
    hx_mlp_kernel<<<128, 256, 0, stream>>>(nodes_elec, h_anti_W, h_anti_b, hxA, 2048);
    hx_mlp_kernel<<< 64, 256, 0, stream>>>(nodes_nuc,  h_ne_W,   h_ne_b,   hxN,  512);

    conv_kernel<<<NBATCH, 128, 0, stream>>>(
        edges_same, edges_anti, edges_ne,
        w_same_W, w_same_b, w_anti_W, w_anti_b, w_ne_W, w_ne_b,
        hxS, hxA, hxN, out_ee, out_ne);
}
